// GAT_4037269258359
// MI455X (gfx1250) — compile-verified, matrix operand order checked by reading, not run on hardware
//
#include <hip/hip_runtime.h>
#include <hip/hip_bf16.h>

// ---------------------------------------------------------------------------
// GAT (2-layer, PyG GATConv semantics) for MI455X / gfx1250.
// GEMMs: v_wmma_f32_16x16x32_bf16 with fragment-prepacked B (one 32B vector
// load per lane per K-step). Segment softmax + aggregation: order-preserving
// uint atomicMax + global_atomic_add_f32 (L2-resident state).
// ---------------------------------------------------------------------------

#define F_IN   256
#define HID    32
#define HEADS  4
#define HC1    128          // HEADS*HID
#define CLS    40
#define CLSP   48           // CLS padded to 3 WMMA n-tiles
#define NEG_SLOPE 0.2f

typedef __attribute__((ext_vector_type(16))) __bf16 v16bf;
typedef __attribute__((ext_vector_type(8)))  float  v8f;
typedef __attribute__((ext_vector_type(4)))  float  v4f;

// ---- helpers --------------------------------------------------------------

// native fp32 -> bf16 (RNE); let the backend pick v_cvt_pk_bf16_f32 etc.
static __device__ __forceinline__ __bf16 f2bf(float f) {
    return static_cast<__bf16>(f);
}

// order-preserving float <-> uint encoding for atomicMax-based segment max
static __device__ __forceinline__ unsigned fenc(float f) {
    unsigned u = __float_as_uint(f);
    return (u & 0x80000000u) ? ~u : (u | 0x80000000u);
}
static __device__ __forceinline__ float fdec(unsigned u) {
    unsigned v = (u & 0x80000000u) ? (u & 0x7FFFFFFFu) : ~u;
    return __uint_as_float(v);
}
#define FENC_NEG_INF 0x007FFFFFu   // fenc(-inf)

static __device__ __forceinline__ void edge_sd(const int* __restrict__ ei,
                                               int E, int e, int& s, int& d) {
    if (e < E) { s = ei[e]; d = ei[E + e]; }
    else       { s = d = e - E; }          // appended self-loops
}

// ---- generic fill ---------------------------------------------------------

__global__ void k_fill_u32(unsigned* __restrict__ p, unsigned v, size_t n) {
    size_t t = (size_t)blockIdx.x * blockDim.x + threadIdx.x;
    if (t < n) p[t] = v;
}

// ---- pack weight matrix into WMMA-B fragment-contiguous bf16 layout -------
// P[((nt*(K/32) + ks)*32 + lane)*16 + j] = bf16(W[k, c]) with
//   klb = (lane>=16)*8 ; k = ks*32 + klb + (j<8 ? j : j+8) ; c = nt*16+lane%16
// so each lane's 16-element fragment is one aligned 32-byte load.

__global__ void k_pack_b(const float* __restrict__ W, __bf16* __restrict__ P,
                         int K, int Nc, int ldw) {
    int t = blockIdx.x * blockDim.x + threadIdx.x;
    int total = (Nc / 16) * (K / 32) * 32 * 16;
    if (t >= total) return;
    int j    = t & 15;
    int lane = (t >> 4) & 31;
    int ks   = (t >> 9) % (K / 32);
    int nt   = (t >> 9) / (K / 32);
    int klb  = (lane >> 4) * 8;
    int koff = (j < 8) ? j : (j + 8);
    int k    = ks * 32 + klb + koff;
    int c    = nt * 16 + (lane & 15);
    float v  = (c < ldw) ? W[(size_t)k * ldw + c] : 0.0f;
    P[t] = f2bf(v);
}

// ---- GEMM1: h1[N,128] = bf16(x[N,256]) @ W1pack ---------------------------
// block = 256 threads = 8 waves; wave w owns n-tile w; grid.x = ceil(N/16).

__global__ void k_gemm1(const float* __restrict__ x, const __bf16* __restrict__ P,
                        float* __restrict__ h1, int N) {
    const int wave = threadIdx.x >> 5;          // 0..7 -> n-tile
    const int lane = threadIdx.x & 31;
    const int mt   = blockIdx.x;
    int row = mt * 16 + (lane & 15);
    if (row >= N) row = N - 1;
    const int col = wave * 16 + (lane & 15);
    const int klb = (lane >> 4) * 8;

    const __bf16* bp = P + ((size_t)wave * (F_IN / 32) * 32 + lane) * 16;
    const float*  xr = x + (size_t)row * F_IN + klb;

    v8f acc = {};
#pragma unroll
    for (int ks = 0; ks < F_IN / 32; ++ks) {
        const v4f* xp4 = (const v4f*)(xr + ks * 32);
        __builtin_prefetch(xr + ks * 32 + 64, 0, 1);
        v4f x0 = xp4[0], x1 = xp4[1];       // K = klb+0..7
        v4f x2 = xp4[4], x3 = xp4[5];       // K = klb+16..23
        v16bf a;
#pragma unroll
        for (int j = 0; j < 4; ++j) { a[j] = f2bf(x0[j]); a[4 + j]  = f2bf(x1[j]); }
#pragma unroll
        for (int j = 0; j < 4; ++j) { a[8 + j] = f2bf(x2[j]); a[12 + j] = f2bf(x3[j]); }
        v16bf b = *(const v16bf*)(bp + (size_t)ks * 32 * 16);
        acc = __builtin_amdgcn_wmma_f32_16x16x32_bf16(
            false, a, false, b, (short)0, acc, false, false);
    }
    // C/D layout: VGPR i -> M = mt*16 + i + 8*(lane>=16), N = lane%16
    const int m0 = mt * 16 + (lane >> 4) * 8;
    float* hp = h1 + (size_t)m0 * HC1 + col;
    if ((mt + 1) * 16 <= N) {
#pragma unroll
        for (int i = 0; i < 8; ++i) hp[(size_t)i * HC1] = acc[i];
    } else {
#pragma unroll
        for (int i = 0; i < 8; ++i)
            if (m0 + i < N) hp[(size_t)i * HC1] = acc[i];
    }
}

// ---- GEMM2: h2[N,48] = helu_bf[N,128] @ W2pack ----------------------------
// block = 96 threads = 3 waves (3 n-tiles of 16 = 48 cols).

__global__ void k_gemm2(const __bf16* __restrict__ A, const __bf16* __restrict__ P,
                        float* __restrict__ h2, int N) {
    const int wave = threadIdx.x >> 5;          // 0..2
    const int lane = threadIdx.x & 31;
    const int mt   = blockIdx.x;
    int row = mt * 16 + (lane & 15);
    if (row >= N) row = N - 1;
    const int col = wave * 16 + (lane & 15);
    const int klb = (lane >> 4) * 8;

    const __bf16* bp = P + ((size_t)wave * (HC1 / 32) * 32 + lane) * 16;
    const __bf16* ar = A + (size_t)row * HC1 + klb;

    v8f acc = {};
#pragma unroll
    for (int ks = 0; ks < HC1 / 32; ++ks) {
        const __bf16* ap = ar + ks * 32;
        v16bf a;
#pragma unroll
        for (int j = 0; j < 8; ++j) { a[j] = ap[j]; a[8 + j] = ap[16 + j]; }
        v16bf b = *(const v16bf*)(bp + (size_t)ks * 32 * 16);
        acc = __builtin_amdgcn_wmma_f32_16x16x32_bf16(
            false, a, false, b, (short)0, acc, false, false);
    }
    const int m0 = mt * 16 + (lane >> 4) * 8;
    float* hp = h2 + (size_t)m0 * CLSP + col;
    if ((mt + 1) * 16 <= N) {
#pragma unroll
        for (int i = 0; i < 8; ++i) hp[(size_t)i * CLSP] = acc[i];
    } else {
#pragma unroll
        for (int i = 0; i < 8; ++i)
            if (m0 + i < N) hp[(size_t)i * CLSP] = acc[i];
    }
}

// ---- per-node attention coefficients --------------------------------------

__global__ void k_alphas1(const float* __restrict__ h1,
                          const float* __restrict__ att_s, const float* __restrict__ att_d,
                          float* __restrict__ aS, float* __restrict__ aD, int N) {
    int t = blockIdx.x * blockDim.x + threadIdx.x;      // N*HEADS
    if (t >= N * HEADS) return;
    int n = t >> 2, hd = t & 3;
    const float* hp = h1 + (size_t)n * HC1 + hd * HID;
    const float* sv = att_s + hd * HID;
    const float* dv = att_d + hd * HID;
    float ss = 0.f, ds = 0.f;
#pragma unroll
    for (int c = 0; c < HID; ++c) { ss += hp[c] * sv[c]; ds += hp[c] * dv[c]; }
    aS[t] = ss; aD[t] = ds;
}

__global__ void k_alphas2(const float* __restrict__ h2,
                          const float* __restrict__ att_s, const float* __restrict__ att_d,
                          float* __restrict__ aS, float* __restrict__ aD, int N) {
    int n = blockIdx.x * blockDim.x + threadIdx.x;
    if (n >= N) return;
    const float* hp = h2 + (size_t)n * CLSP;
    float ss = 0.f, ds = 0.f;
#pragma unroll
    for (int c = 0; c < CLS; ++c) { ss += hp[c] * att_s[c]; ds += hp[c] * att_d[c]; }
    aS[n] = ss; aD[n] = ds;
}

// ---- segment softmax: pass 1 (max), pass 2 (exp + denom) ------------------

__global__ void k_att_max(const int* __restrict__ ei, int E, int Etot, int H,
                          const float* __restrict__ aS, const float* __restrict__ aD,
                          unsigned* __restrict__ mx) {
    size_t t = (size_t)blockIdx.x * blockDim.x + threadIdx.x;
    if (t >= (size_t)Etot * H) return;
    int e = (int)(t / H), hd = (int)(t % H);
    int s, d; edge_sd(ei, E, e, s, d);
    float v = aS[(size_t)s * H + hd] + aD[(size_t)d * H + hd];
    v = v > 0.f ? v : NEG_SLOPE * v;                     // leaky_relu
    atomicMax(&mx[(size_t)d * H + hd], fenc(v));
}

__global__ void k_att_exp(const int* __restrict__ ei, int E, int Etot, int H,
                          const float* __restrict__ aS, const float* __restrict__ aD,
                          const unsigned* __restrict__ mx,
                          float* __restrict__ exbuf, float* __restrict__ den) {
    size_t t = (size_t)blockIdx.x * blockDim.x + threadIdx.x;
    if (t >= (size_t)Etot * H) return;
    int e = (int)(t / H), hd = (int)(t % H);
    int s, d; edge_sd(ei, E, e, s, d);
    float v = aS[(size_t)s * H + hd] + aD[(size_t)d * H + hd];
    v = v > 0.f ? v : NEG_SLOPE * v;
    float ex = __expf(v - fdec(mx[(size_t)d * H + hd]));
    exbuf[t] = ex;
    atomicAdd(&den[(size_t)d * H + hd], ex);
}

// ---- weighted scatter-aggregate: out[dst] += h[src] * alpha ---------------
// One thread per (edge, 4 consecutive channels): float4 gather + 4 atomics.

__global__ void k_aggregate4(const int* __restrict__ ei, int E, int Etot,
                             const float* __restrict__ hfeat, int hstride,
                             int Cper, int H,
                             const float* __restrict__ exbuf,
                             const float* __restrict__ den,
                             float* __restrict__ out, int ostride) {
    const int HC = H * Cper;
    const int Q  = HC >> 2;                 // 4-wide chunks per edge
    size_t t = (size_t)blockIdx.x * blockDim.x + threadIdx.x;
    if (t >= (size_t)Etot * Q) return;
    int e  = (int)(t / Q);
    int r  = (int)(t % Q) << 2;
    int hd = r / Cper;
    int s, d; edge_sd(ei, E, e, s, d);
    float alpha = exbuf[(size_t)e * H + hd] /
                  (den[(size_t)d * H + hd] + 1e-16f);
    v4f hv = *(const v4f*)(hfeat + (size_t)s * hstride + r);
    float* op = out + (size_t)d * ostride + r;
    atomicAdd(op + 0, hv[0] * alpha);
    atomicAdd(op + 1, hv[1] * alpha);
    atomicAdd(op + 2, hv[2] * alpha);
    atomicAdd(op + 3, hv[3] * alpha);
}

// ---- bias + ELU (layer 1), bias (layer 2) ---------------------------------

__global__ void k_bias_elu(float* __restrict__ agg, const float* __restrict__ b1,
                           __bf16* __restrict__ hbf, size_t total) {
    size_t t = (size_t)blockIdx.x * blockDim.x + threadIdx.x;
    if (t >= total) return;
    int r = (int)(t % HC1);
    float v = agg[t] + b1[r];
    v = v > 0.f ? v : expm1f(v);                         // ELU, alpha=1
    agg[t] = v;
    hbf[t] = f2bf(v);
}

__global__ void k_add_bias(float* __restrict__ out, const float* __restrict__ b,
                           int C, size_t total) {
    size_t t = (size_t)blockIdx.x * blockDim.x + threadIdx.x;
    if (t >= total) return;
    out[t] += b[(int)(t % C)];
}

// ---------------------------------------------------------------------------

extern "C" void kernel_launch(void* const* d_in, const int* in_sizes, int n_in,
                              void* d_out, int out_size, void* d_ws, size_t ws_size,
                              hipStream_t stream) {
    const int*   ei   = (const int*)  d_in[0];   // [2,E]
    const float* x    = (const float*)d_in[1];   // [N,256]
    const float* W1   = (const float*)d_in[2];   // [256,128]
    const float* as1  = (const float*)d_in[3];   // [4,32]
    const float* ad1  = (const float*)d_in[4];
    const float* b1   = (const float*)d_in[5];   // [128]
    const float* W2   = (const float*)d_in[6];   // [128,40]
    const float* as2  = (const float*)d_in[7];   // [1,40]
    const float* ad2  = (const float*)d_in[8];
    const float* b2   = (const float*)d_in[9];   // [40]
    float* out = (float*)d_out;                  // [N,40]

    const int E    = in_sizes[0] / 2;
    const int N    = in_sizes[1] / F_IN;
    const int Etot = E + N;                      // + self-loops
    const int MT   = (N + 15) / 16;

    // ---- workspace layout (256B aligned) ----
    char* w = (char*)d_ws;
    auto take = [&](size_t bytes) -> char* {
        char* p = w; w += (bytes + 255) & ~(size_t)255; return p;
    };
    __bf16* W1p   = (__bf16*)take(sizeof(__bf16) * F_IN * HC1);
    __bf16* W2p   = (__bf16*)take(sizeof(__bf16) * HC1 * CLSP);
    float*  h1    = (float*) take(sizeof(float) * (size_t)N * HC1);
    float*  aS1   = (float*) take(sizeof(float) * (size_t)N * HEADS);
    float*  aD1   = (float*) take(sizeof(float) * (size_t)N * HEADS);
    unsigned* mx1 = (unsigned*)take(sizeof(unsigned) * (size_t)N * HEADS);
    float*  den1  = (float*) take(sizeof(float) * (size_t)N * HEADS);
    float*  ex1   = (float*) take(sizeof(float) * (size_t)Etot * HEADS);
    float*  agg1  = (float*) take(sizeof(float) * (size_t)N * HC1);  // -> h_elu
    __bf16* heb   = (__bf16*)take(sizeof(__bf16) * (size_t)N * HC1);
    float*  h2    = (float*) take(sizeof(float) * (size_t)N * CLSP);
    float*  aS2   = (float*) take(sizeof(float) * (size_t)N);
    float*  aD2   = (float*) take(sizeof(float) * (size_t)N);
    unsigned* mx2 = (unsigned*)take(sizeof(unsigned) * (size_t)N);
    float*  den2  = (float*) take(sizeof(float) * (size_t)N);
    float*  ex2   = (float*) take(sizeof(float) * (size_t)Etot);

    const int TB = 256;
    auto blk = [&](size_t n) -> unsigned { return (unsigned)((n + TB - 1) / TB); };

    // ---- layer 1 ----
    k_pack_b<<<blk(F_IN * HC1), TB, 0, stream>>>(W1, W1p, F_IN, HC1, HC1);
    k_pack_b<<<blk(HC1 * CLSP), TB, 0, stream>>>(W2, W2p, HC1, CLSP, CLS);

    k_gemm1<<<MT, 256, 0, stream>>>(x, W1p, h1, N);

    k_alphas1<<<blk((size_t)N * HEADS), TB, 0, stream>>>(h1, as1, ad1, aS1, aD1, N);

    k_fill_u32<<<blk((size_t)N * HEADS), TB, 0, stream>>>(mx1, FENC_NEG_INF, (size_t)N * HEADS);
    k_fill_u32<<<blk((size_t)N * HEADS), TB, 0, stream>>>((unsigned*)den1, 0u, (size_t)N * HEADS);
    k_fill_u32<<<blk((size_t)N * HC1), TB, 0, stream>>>((unsigned*)agg1, 0u, (size_t)N * HC1);

    k_att_max<<<blk((size_t)Etot * HEADS), TB, 0, stream>>>(ei, E, Etot, HEADS, aS1, aD1, mx1);
    k_att_exp<<<blk((size_t)Etot * HEADS), TB, 0, stream>>>(ei, E, Etot, HEADS, aS1, aD1, mx1, ex1, den1);
    k_aggregate4<<<blk((size_t)Etot * (HC1 / 4)), TB, 0, stream>>>(
        ei, E, Etot, h1, HC1, HID, HEADS, ex1, den1, agg1, HC1);

    k_bias_elu<<<blk((size_t)N * HC1), TB, 0, stream>>>(agg1, b1, heb, (size_t)N * HC1);

    // ---- layer 2 ----
    k_gemm2<<<MT, 96, 0, stream>>>(heb, W2p, h2, N);

    k_alphas2<<<blk((size_t)N), TB, 0, stream>>>(h2, as2, ad2, aS2, aD2, N);

    k_fill_u32<<<blk((size_t)N), TB, 0, stream>>>(mx2, FENC_NEG_INF, (size_t)N);
    k_fill_u32<<<blk((size_t)N), TB, 0, stream>>>((unsigned*)den2, 0u, (size_t)N);
    k_fill_u32<<<blk((size_t)N * CLS), TB, 0, stream>>>((unsigned*)out, 0u, (size_t)N * CLS);

    k_att_max<<<blk((size_t)Etot), TB, 0, stream>>>(ei, E, Etot, 1, aS2, aD2, mx2);
    k_att_exp<<<blk((size_t)Etot), TB, 0, stream>>>(ei, E, Etot, 1, aS2, aD2, mx2, ex2, den2);
    k_aggregate4<<<blk((size_t)Etot * (CLS / 4)), TB, 0, stream>>>(
        ei, E, Etot, h2, CLSP, CLS, 1, ex2, den2, out, CLS);

    k_add_bias<<<blk((size_t)N * CLS), TB, 0, stream>>>(out, b2, CLS, (size_t)N * CLS);
}
